// Memory_5669356835754
// MI455X (gfx1250) — matrix operand order repaired and checked by reading, not compiled, not run on hardware
//
#include <hip/hip_runtime.h>

typedef __attribute__((ext_vector_type(16))) __bf16       v16bf;
typedef __attribute__((ext_vector_type(8)))  float        v8f;
typedef __attribute__((ext_vector_type(4)))  unsigned int u32x4;
typedef __attribute__((ext_vector_type(4)))  int          i32x4;
typedef __attribute__((address_space(1)))    i32x4        gas_i32x4;  // global
typedef __attribute__((address_space(3)))    i32x4        las_i32x4;  // LDS

union ABFrag { u32x4 u[2]; v16bf v; };

#define N_ADDR   68
#define CCH      256
#define HWP      256                 // 16x16 spatial
#define M_TOTAL  (N_ADDR * HWP)      // 17408 GEMM rows
#define XELEMS   ((size_t)N_ADDR * CCH * HWP)   // 68*65536 elements
#define WELEMS   ((size_t)9 * CCH * CCH)        // 589824 elements
#define NTILES   72                  // 9 (kh,kw) * 8 ci-tiles of 32

// CDNA5 async global->LDS path (ASYNCcnt), with portable fallback.
#if __has_builtin(__builtin_amdgcn_global_load_async_to_lds_b128)
#define CDNA5_ASYNC_LDS 1
#endif

__device__ __forceinline__ void wait_async_lds() {
#ifdef CDNA5_ASYNC_LDS
#if __has_builtin(__builtin_amdgcn_s_wait_asynccnt)
    __builtin_amdgcn_s_wait_asynccnt(0);
#else
    asm volatile("s_wait_asynccnt 0" ::: "memory");
#endif
#endif
}

// ---------------------------------------------------------------------------
// Dynamic memory: segment mean over keys, written directly into d_out.
// Deterministic: single thread builds the member list in fixed order.
// ---------------------------------------------------------------------------
__global__ __launch_bounds__(256)
void mem_mean(const int* __restrict__ style, const int* __restrict__ comp,
              const float* __restrict__ feats, float* __restrict__ out)
{
    __shared__ int s_m[768];
    __shared__ int s_cnt;
    const int p   = blockIdx.x;          // pair index b*3+j, 0..767
    const int tid = threadIdx.x;
    if (tid == 0) {
        const int offc[3] = {0, 19, 40};
        const int keyp = style[p / 3] * 68 + comp[p] + offc[p % 3];
        int cnt = 0;
        for (int q = 0; q < 768; ++q) {
            const int kq = style[q / 3] * 68 + comp[q] + offc[q % 3];
            if (kq == keyp) s_m[cnt++] = q;
        }
        s_cnt = cnt;
    }
    __syncthreads();
    const int  cnt = s_cnt;              // >= 1 (p is always a member)
    const long eo  = (long)blockIdx.y * 1024 + tid * 4;
    float4 acc = make_float4(0.f, 0.f, 0.f, 0.f);
    for (int i = 0; i < cnt; ++i) {
        const float4 v = *(const float4*)(feats + (long)s_m[i] * 65536 + eo);
        acc.x += v.x; acc.y += v.y; acc.z += v.z; acc.w += v.w;
    }
    const float inv = 1.0f / (float)cnt;
    acc.x *= inv; acc.y *= inv; acc.z *= inv; acc.w *= inv;
    *(float4*)(out + (long)p * 65536 + eo) = acc;
}

// ---------------------------------------------------------------------------
// Stage-1 input prep: relu, fp32 -> bf16 hi/lo split, NCHW -> NHWC.
// ---------------------------------------------------------------------------
__global__ __launch_bounds__(256)
void prep_input(const float* __restrict__ pm, __bf16* __restrict__ xh,
                __bf16* __restrict__ xl)
{
    const long idx = (long)blockIdx.x * 256 + threadIdx.x;   // [a][hw][c]
    const int  a   = (int)(idx >> 16);
    const int  rem = (int)(idx & 65535);
    const int  hw  = rem >> 8;
    const int  c   = rem & 255;
    const float v  = fmaxf(pm[(long)a * 65536 + (long)c * 256 + hw], 0.0f);
    const __bf16 hi = (__bf16)v;
    xh[idx] = hi;
    xl[idx] = (__bf16)(v - (float)hi);
}

// OIHW w[co][ci][kh][kw] -> [kk][co][ci] bf16 hi/lo (ci contiguous for B frags)
__global__ __launch_bounds__(256)
void prep_weights(const float* __restrict__ w, __bf16* __restrict__ wh,
                  __bf16* __restrict__ wl)
{
    const int idx = blockIdx.x * 256 + threadIdx.x;          // [kk][co][ci]
    const int kk  = idx >> 16;
    const int rem = idx & 65535;
    const int co  = rem >> 8;
    const int ci  = rem & 255;
    const float v = w[(long)co * 2304 + ci * 9 + kk];
    const __bf16 hi = (__bf16)v;
    wh[idx] = hi;
    wl[idx] = (__bf16)(v - (float)hi);
}

// ---------------------------------------------------------------------------
// Implicit-GEMM 3x3 conv via v_wmma_f32_16x16x32_bf16 with hi/lo fp32 split.
//   M = (a,h,w) rows (17408), N = co (256), K = (kh,kw,ci) (2304)
// Block: 256 thr = 8 waves; block tile M=256 (one image) x N=64.
// Wave tile 32x64: 8 accumulators; A frag reused over 4 N-subtiles,
// B frags reused over 2 M-subtiles and read from an LDS tile that is
// double-buffered via GLOBAL_LOAD_ASYNC_TO_LDS_B128 (ASYNCcnt).
// ---------------------------------------------------------------------------
__global__ __launch_bounds__(256)
void conv_wmma(const __bf16* __restrict__ xh, const __bf16* __restrict__ xl,
               const __bf16* __restrict__ wh, const __bf16* __restrict__ wl,
               const float* __restrict__ bias,
               __bf16* __restrict__ oxh, __bf16* __restrict__ oxl,
               float* __restrict__ ohyp, int final_stage)
{
    __shared__ __bf16 sB[2][2][2048];   // [buf][hi/lo][co*32+ci] : 16 KB

    const int tid  = threadIdx.x;
    const int wave = tid >> 5;
    const int lane = tid & 31;
    const int half = lane >> 4;          // K-half select per ISA layout
    const int r    = lane & 15;

    const int a      = blockIdx.x;       // one 16x16 image per block
    const int n_base = blockIdx.y * 64;
    const int h0     = wave << 1;        // h = wave*2 + sub, w = r

    // Cooperative B-tile load mapping: thread -> one 16B chunk per hi/lo
    const int e0  = tid * 8;             // element in [co(6b)][ci(5b)] tile
    const int bco = e0 >> 5;
    const int bci = e0 & 31;

    auto issue_tile = [&](int t, int buf) {
        const int  kk  = t >> 3;
        const int  ci0 = (t & 7) << 5;
        const long g   = ((long)kk << 16) + (long)(n_base + bco) * 256 + ci0 + bci;
#ifdef CDNA5_ASYNC_LDS
        __builtin_amdgcn_global_load_async_to_lds_b128(
            (gas_i32x4*)(wh + g), (las_i32x4*)&sB[buf][0][e0], 0, 0);
        __builtin_amdgcn_global_load_async_to_lds_b128(
            (gas_i32x4*)(wl + g), (las_i32x4*)&sB[buf][1][e0], 0, 0);
#else
        *(u32x4*)&sB[buf][0][e0] = *(const u32x4*)(wh + g);
        *(u32x4*)&sB[buf][1][e0] = *(const u32x4*)(wl + g);
#endif
    };

    v8f acc[2][4] = {};

    issue_tile(0, 0);
    for (int t = 0; t < NTILES; ++t) {
        const int buf = t & 1;
        // Own async loads done, then rendezvous: tile `buf` ready for all,
        // and every wave has finished computing on buf^1 -> safe to refill.
        wait_async_lds();
        __syncthreads();
        if (t + 1 < NTILES) issue_tile(t + 1, buf ^ 1);

        const int kk  = t >> 3;
        const int kh  = kk / 3, kw = kk % 3;
        const int ci0 = (t & 7) << 5;
        const int wi_ = r + kw - 1;
        const bool wok = (unsigned)wi_ < 16u;

        // A fragments: lane holds K = {0..7,16..23} + half*8 (ISA 16-bit A)
        ABFrag ah[2], al[2];
#pragma unroll
        for (int sub = 0; sub < 2; ++sub) {
            const int  hh    = h0 + sub + kh - 1;
            const bool valid = wok && ((unsigned)hh < 16u);
            ABFrag AH{}, AL{};
            if (valid) {
                const long arow = (long)((a << 8) + (hh << 4) + wi_) << 8;
                const int  c0   = ci0 + half * 8;
                AH.u[0] = *(const u32x4*)(xh + arow + c0);
                AH.u[1] = *(const u32x4*)(xh + arow + c0 + 16);
                AL.u[0] = *(const u32x4*)(xl + arow + c0);
                AL.u[1] = *(const u32x4*)(xl + arow + c0 + 16);
            }
            ah[sub] = AH;
            al[sub] = AL;
        }

#pragma unroll
        for (int j = 0; j < 4; ++j) {
            // B fragment from LDS: lane = N column, K = {0..7,16..23} + half*8
            const int cb = (j * 16 + r) * 32 + half * 8;
            ABFrag bh, bl;
            bh.u[0] = *(const u32x4*)&sB[buf][0][cb];
            bh.u[1] = *(const u32x4*)&sB[buf][0][cb + 16];
            bl.u[0] = *(const u32x4*)&sB[buf][1][cb];
            bl.u[1] = *(const u32x4*)&sB[buf][1][cb + 16];
#pragma unroll
            for (int sub = 0; sub < 2; ++sub) {
                // fp32-accurate split product: hi*hi + hi*lo + lo*hi
                acc[sub][j] = __builtin_amdgcn_wmma_f32_16x16x32_bf16(
                    false, ah[sub].v, false, bh.v, (short)0, acc[sub][j], false, false);
                acc[sub][j] = __builtin_amdgcn_wmma_f32_16x16x32_bf16(
                    false, ah[sub].v, false, bl.v, (short)0, acc[sub][j], false, false);
                acc[sub][j] = __builtin_amdgcn_wmma_f32_16x16x32_bf16(
                    false, al[sub].v, false, bh.v, (short)0, acc[sub][j], false, false);
            }
        }
    }

    // Epilogue. D layout: VGPR i -> row = tile_m + half*8 + i, col = n0 + r
#pragma unroll
    for (int j = 0; j < 4; ++j) {
        const int   co = n_base + j * 16 + r;
        const float bv = bias[co];
#pragma unroll
        for (int sub = 0; sub < 2; ++sub) {
            const int hwrow = (wave << 5) + (sub << 4) + (half << 3);
#pragma unroll
            for (int i = 0; i < 8; ++i) {
                const int   hw2 = hwrow + i;
                const float val = acc[sub][j][i] + bv;
                if (final_stage) {
                    ohyp[(long)a * 65536 + (long)co * 256 + hw2] = val;  // NCHW
                } else {
                    const float  rv = fmaxf(val, 0.0f);  // next stage input relu
                    const __bf16 hi = (__bf16)rv;
                    const long   o  = ((long)a * 256 + hw2) * 256 + co;  // NHWC
                    oxh[o] = hi;
                    oxl[o] = (__bf16)(rv - (float)hi);
                }
            }
        }
    }
}

// ---------------------------------------------------------------------------
// Final gather-add: out[b,j] += hyp[addr(b,j)]
// ---------------------------------------------------------------------------
__global__ __launch_bounds__(256)
void add_hyp(const int* __restrict__ comp, const float* __restrict__ hyp,
             float* __restrict__ out)
{
    const int p       = blockIdx.x;
    const int offc[3] = {0, 19, 40};
    const int addr    = comp[p] + offc[p % 3];
    const long eo     = (long)blockIdx.y * 1024 + threadIdx.x * 4;
    const float4 hv   = *(const float4*)(hyp + (long)addr * 65536 + eo);
    float4 ov         = *(float4*)(out + (long)p * 65536 + eo);
    ov.x += hv.x; ov.y += hv.y; ov.z += hv.z; ov.w += hv.w;
    *(float4*)(out + (long)p * 65536 + eo) = ov;
}

extern "C" void kernel_launch(void* const* d_in, const int* in_sizes, int n_in,
                              void* d_out, int out_size, void* d_ws, size_t ws_size,
                              hipStream_t stream)
{
    (void)in_sizes; (void)n_in; (void)out_size; (void)ws_size;
    const int*   style = (const int*)d_in[0];
    const int*   comp  = (const int*)d_in[1];
    const float* feats = (const float*)d_in[2];
    const float* pm    = (const float*)d_in[3];
    const float* wcv[3] = {(const float*)d_in[4], (const float*)d_in[6], (const float*)d_in[8]};
    const float* bcv[3] = {(const float*)d_in[5], (const float*)d_in[7], (const float*)d_in[9]};
    float* out = (float*)d_out;

    // Workspace carve-up (~61 MB)
    char*  ws  = (char*)d_ws;
    size_t off = 0;
    auto alloc = [&](size_t bytes) {
        char* p = ws + off;
        off += (bytes + 255) & ~(size_t)255;
        return p;
    };
    __bf16* xh0 = (__bf16*)alloc(XELEMS * 2);
    __bf16* xl0 = (__bf16*)alloc(XELEMS * 2);
    __bf16* xh1 = (__bf16*)alloc(XELEMS * 2);
    __bf16* xl1 = (__bf16*)alloc(XELEMS * 2);
    __bf16* whv[3], *wlv[3];
    for (int s = 0; s < 3; ++s) {
        whv[s] = (__bf16*)alloc(WELEMS * 2);
        wlv[s] = (__bf16*)alloc(WELEMS * 2);
    }
    float* hyp = (float*)alloc(XELEMS * 4);

    // 1) dynamic-memory segment mean -> d_out
    mem_mean<<<dim3(768, 64), 256, 0, stream>>>(style, comp, feats, out);

    // 2) weight + input prep (relu, hi/lo bf16 split, GEMM-friendly layouts)
    for (int s = 0; s < 3; ++s)
        prep_weights<<<2304, 256, 0, stream>>>(wcv[s], whv[s], wlv[s]);
    prep_input<<<17408, 256, 0, stream>>>(pm, xh0, xl0);

    // 3) hypernet: 3 WMMA implicit-GEMM convs on the 68 unique addresses
    const dim3 cgrid(N_ADDR, CCH / 64);
    conv_wmma<<<cgrid, 256, 0, stream>>>(xh0, xl0, whv[0], wlv[0], bcv[0],
                                         xh1, xl1, nullptr, 0);
    conv_wmma<<<cgrid, 256, 0, stream>>>(xh1, xl1, whv[1], wlv[1], bcv[1],
                                         xh0, xl0, nullptr, 0);
    conv_wmma<<<cgrid, 256, 0, stream>>>(xh0, xl0, whv[2], wlv[2], bcv[2],
                                         nullptr, nullptr, hyp, 1);

    // 4) out[b,j] += hyp[addr(b,j)]
    add_hyp<<<dim3(768, 64), 256, 0, stream>>>(comp, hyp, out);
}
